// DIM_86870008529601
// MI455X (gfx1250) — matrix operand order repaired and checked by reading, not compiled, not run on hardware
//
#include <hip/hip_runtime.h>
#include <hip/hip_bf16.h>

// ---------------------------------------------------------------------------
// Types for gfx1250 WMMA (wave32): V_WMMA_F32_16X16X32_BF16
// ---------------------------------------------------------------------------
typedef __attribute__((ext_vector_type(16))) __bf16 v16bf;
typedef __attribute__((ext_vector_type(8)))  __bf16 v8bf;
typedef __attribute__((ext_vector_type(8)))  float  v8f;

__device__ inline v8f wmma_bf16(v16bf a, v16bf b, v8f c) {
  // 8 args: (neg_a, A, neg_b, B, c_mod, C, reuse_a, reuse_b)
  return __builtin_amdgcn_wmma_f32_16x16x32_bf16(false, a, false, b, (short)0, c,
                                                 false, false);
}

__device__ inline float fsig(float x)  { return 1.0f / (1.0f + __expf(-x)); }
__device__ inline float ftanh(float x) { return 1.0f - 2.0f / (1.0f + __expf(2.0f * x)); }

// A-operand 16x32 bf16 layout: lane m = lane&15; low lanes K = c*32+{0..7,16..23},
// high lanes +8.  B-operand 32x16: lane n = lane&15; same K order as A.
__device__ inline v16bf ldsPackA(const __bf16* mat, int row, int c, bool hi) {
  int k1 = c * 32 + (hi ? 8 : 0);
  const __bf16* p = mat + row * 128 + k1;
  v8bf lo = *(const v8bf*)p;
  v8bf h2 = *(const v8bf*)(p + 16);
  v16bf r;
#pragma unroll
  for (int e = 0; e < 8; ++e) { r[e] = lo[e]; r[8 + e] = h2[e]; }
  return r;
}
__device__ inline v16bf ldsPackB(const __bf16* mat, int row, int c, bool hi) {
  const __bf16* p = mat + row * 128 + c * 32 + (hi ? 8 : 0);
  v8bf lo = *(const v8bf*)p;
  v8bf h2 = *(const v8bf*)(p + 16);
  v16bf r;
#pragma unroll
  for (int e = 0; e < 8; ++e) { r[e] = lo[e]; r[8 + e] = h2[e]; }
  return r;
}

// ---------------------------------------------------------------------------
// Generic bf16-WMMA GEMM: Out[m,n] = alpha * sum_k A[m,k] * B(k,n) + bias[n]
//   TRANS=true : B(k,n) = W[n*ldw + k]   (x @ W.T with row-major W)
//   TRANS=false: B(k,n) = W[k*ldw + n]
// One 16-row M-strip per wave, NT column tiles of 16, K zero-padded to 32.
// ---------------------------------------------------------------------------
template<int NT, bool TRANS>
__global__ __launch_bounds__(256) void gemm_wmma(
    const float* __restrict__ A, int M, int K, int lda,
    const float* __restrict__ W, int ldw, int N,
    const float* __restrict__ bias, float alpha,
    float* __restrict__ Out, int ldo)
{
  const int lane  = threadIdx.x & 31;
  const int wave  = threadIdx.x >> 5;
  const int mtile = blockIdx.x * (blockDim.x >> 5) + wave;
  if (mtile * 16 >= M) return;              // wave-uniform exit (EXEC stays all-1 for WMMA)
  const int  ln  = lane & 15;
  const bool hi  = lane >= 16;
  const int  mA  = mtile * 16 + ln;
  const bool mok = mA < M;

  v8f acc[NT] = {};
  const int KC = (K + 31) >> 5;
  for (int c = 0; c < KC; ++c) {
    const int kbA = c * 32 + (hi ? 8 : 0);
    v16bf a;
#pragma unroll
    for (int e = 0; e < 16; ++e) {
      int k = kbA + ((e < 8) ? e : (8 + e));
      float v = (mok && k < K) ? A[(long)mA * lda + k] : 0.0f;
      a[e] = (__bf16)v;
    }
    const int kbB = c * 32 + (hi ? 8 : 0);
#pragma unroll
    for (int j = 0; j < NT; ++j) {
      int n = j * 16 + ln;
      bool nok = n < N;
      v16bf b;
#pragma unroll
      for (int e = 0; e < 16; ++e) {
        int k = kbB + ((e < 8) ? e : (8 + e));
        float v = 0.0f;
        if (nok && k < K) v = TRANS ? W[(long)n * ldw + k] : W[(long)k * ldw + n];
        b[e] = (__bf16)v;
      }
      acc[j] = wmma_bf16(a, b, acc[j]);
    }
  }
#pragma unroll
  for (int j = 0; j < NT; ++j) {
    int n = j * 16 + ln;
    if (n < N) {
      float bv = bias ? bias[n] : 0.0f;
#pragma unroll
      for (int r = 0; r < 8; ++r) {
        int m = mtile * 16 + r + (hi ? 8 : 0);
        if (m < M) Out[(long)m * ldo + n] = alpha * acc[j][r] + bv;
      }
    }
  }
}

// ---------------------------------------------------------------------------
// Small constant tables: cvec_* = bias + skill_b @ W_sk.T ; ANS* = answer rows
// through the second column-block of Wo/Wq/W1 (first H output rows only).
// ---------------------------------------------------------------------------
__global__ void misc_kernel(
    const float* __restrict__ Wip, const float* __restrict__ bip,
    const float* __restrict__ skill_b,
    const float* __restrict__ Wtp, const float* __restrict__ btp,
    const float* __restrict__ tskill_b,
    const float* __restrict__ Wo, const float* __restrict__ bo,
    const float* __restrict__ Wq, const float* __restrict__ bq,
    const float* __restrict__ W1, const float* __restrict__ b1,
    const float* __restrict__ ans_emb,
    float* __restrict__ cvec_ip, float* __restrict__ cvec_tp,
    float* __restrict__ ANSo, float* __restrict__ ANSq, float* __restrict__ ANS1)
{
  int n = threadIdx.x;
  if (n >= 128) return;
  float s1 = bip[n], s2 = btp[n];
  for (int k = 0; k < 128; ++k) {
    s1 += skill_b[k]  * Wip[n * 512 + 128 + k];
    s2 += tskill_b[k] * Wtp[n * 512 + 128 + k];
  }
  cvec_ip[n] = s1; cvec_tp[n] = s2;
  for (int a = 0; a < 2; ++a) {
    float so = bo[n], sq = bq[n], sv = b1[n];
    for (int k = 0; k < 128; ++k) {
      float ae = ans_emb[a * 128 + k];
      so += ae * Wo[n * 256 + 128 + k];
      sq += ae * Wq[n * 256 + 128 + k];
      sv += ae * W1[n * 512 + 128 + k];
    }
    ANSo[a * 128 + n] = so; ANSq[a * 128 + n] = sq; ANS1[a * 128 + n] = sv;
  }
}

// data[m,:] += P[p[m]] + SD[sd[m]] + D[d[m]]   (tables already carry the biases)
__global__ void gather_add_kernel(
    float* __restrict__ data,
    const float* __restrict__ P, const float* __restrict__ SD,
    const float* __restrict__ D,
    const int* __restrict__ ip, const int* __restrict__ isd,
    const int* __restrict__ idd, long rows)
{
  long i = (long)blockIdx.x * blockDim.x + threadIdx.x;
  if (i >= rows * 128) return;
  int m = (int)(i >> 7), n = (int)(i & 127);
  data[i] += P[(long)ip[m] * 128 + n] + SD[(long)isd[m] * 128 + n] +
             D[(long)idd[m] * 128 + n];
}

// ---------------------------------------------------------------------------
// Recurrence: one wave per 16 batch rows, weights (5 x HxH bf16) resident in
// LDS, kkk ping-ponged LDS<->WMMA layouts.  A-operand fragments are hoisted
// out of the N-tile loops (loaded once per step per stage, reused 8x) so each
// WMMA only waits on its own two B ds_load_b128's.
// ---------------------------------------------------------------------------
__global__ __launch_bounds__(32) void rnn_wmma(
    const float* __restrict__ Wl, const float* __restrict__ Wc,
    const float* __restrict__ Wo, const float* __restrict__ Wq,
    const float* __restrict__ W1,
    const float* __restrict__ prel, const float* __restrict__ prec,
    const float* __restrict__ ANSo, const float* __restrict__ ANSq,
    const float* __restrict__ ANS1,
    const float* __restrict__ SD1, const float* __restrict__ D1,
    const int* __restrict__ x_answer, const int* __restrict__ input_sd,
    const int* __restrict__ input_d,
    const float* __restrict__ target_data,
    const float* __restrict__ kw,
    float* __restrict__ logits)
{
  extern __shared__ char smemraw[];
  __bf16* WlS  = (__bf16*)smemraw;
  __bf16* WcS  = WlS + 16384;
  __bf16* WoS  = WcS + 16384;
  __bf16* WqS  = WoS + 16384;
  __bf16* W1S  = WqS + 16384;
  __bf16* kkkA = W1S + 16384;   // 16x128 bf16 (A layout source)
  __bf16* cccA = kkkA + 2048;   // 16x128 bf16
  float*  kkkF = (float*)(cccA + 2048);  // 16x128 f32
  int*    idxS = (int*)(kkkF + 2048);    // a[16], sd[16], d[16]

  const int  tid  = threadIdx.x;
  const int  lane = tid & 31;
  const int  ln   = lane & 15;
  const bool hi   = lane >= 16;
  const int  b0   = blockIdx.x * 16;

  // Stage weights (first 128 output rows, relevant column blocks) into LDS.
  for (int i = tid; i < 16384; i += 32) {
    int n = i >> 7, k = i & 127;
    WlS[i] = (__bf16)Wl[n * 128 + k];
    WcS[i] = (__bf16)Wc[n * 128 + k];
    WoS[i] = (__bf16)Wo[n * 256 + k];
    WqS[i] = (__bf16)Wq[n * 256 + k];
    W1S[i] = (__bf16)W1[n * 512 + k];
  }
  for (int i = tid; i < 2048; i += 32) {
    float v = kw[i & 127];
    kkkF[i] = v;
    kkkA[i] = (__bf16)v;
  }
  __syncthreads();

  for (int t = 0; t < 200; ++t) {
    if (tid < 16) {
      int g = (b0 + tid) * 200 + t;
      idxS[tid]      = x_answer[g];
      idxS[16 + tid] = input_sd[g];
      idxS[32 + tid] = input_d[g];
    }
    __syncthreads();

    // ---- stage 1: gate/c/o1 matmuls from kkk --------------------------------
    v16bf Ak[4];
#pragma unroll
    for (int c = 0; c < 4; ++c) Ak[c] = ldsPackA(kkkA, ln, c, hi);

    v8f o1s[8];
#pragma unroll
    for (int j = 0; j < 8; ++j) {
      v8f aL = {}, aC = {}, a1 = {};
#pragma unroll
      for (int c = 0; c < 4; ++c) {
        aL = wmma_bf16(Ak[c], ldsPackB(WlS, j * 16 + ln, c, hi), aL);
        aC = wmma_bf16(Ak[c], ldsPackB(WcS, j * 16 + ln, c, hi), aC);
        a1 = wmma_bf16(Ak[c], ldsPackB(W1S, j * 16 + ln, c, hi), a1);
      }
      int n = j * 16 + ln;
#pragma unroll
      for (int r = 0; r < 8; ++r) {
        int m = r + (hi ? 8 : 0);
        long g = ((long)(b0 + m) * 200 + t) * 128 + n;
        float gate = fsig(aL[r] + prel[g]);
        float cc   = ftanh(aC[r] + prec[g]);
        float p1   = ANS1[idxS[m] * 128 + n] + SD1[idxS[16 + m] * 128 + n] +
                     D1[idxS[32 + m] * 128 + n];
        o1s[j][r] = fsig(a1[r] + p1);
        cccA[m * 128 + n] = (__bf16)(gate * cc);
      }
    }
    __syncthreads();

    // ---- stage 2: xx matmuls from ccc, state update -------------------------
    v16bf Ac[4];
#pragma unroll
    for (int c = 0; c < 4; ++c) Ac[c] = ldsPackA(cccA, ln, c, hi);

#pragma unroll
    for (int j = 0; j < 8; ++j) {
      v8f aO = {}, aQ = {};
#pragma unroll
      for (int c = 0; c < 4; ++c) {
        aO = wmma_bf16(Ac[c], ldsPackB(WoS, j * 16 + ln, c, hi), aO);
        aQ = wmma_bf16(Ac[c], ldsPackB(WqS, j * 16 + ln, c, hi), aQ);
      }
      int n = j * 16 + ln;
#pragma unroll
      for (int r = 0; r < 8; ++r) {
        int m = r + (hi ? 8 : 0);
        int av = idxS[m];
        float xx = fsig(aO[r] + ANSo[av * 128 + n]) *
                   ftanh(aQ[r] + ANSq[av * 128 + n]);
        float o1 = o1s[j][r];
        float kn = o1 * kkkF[m * 128 + n] + (1.0f - o1) * xx;
        kkkF[m * 128 + n] = kn;
        kkkA[m * 128 + n] = (__bf16)kn;
      }
    }
    __syncthreads();

    // ---- logits for this timestep ------------------------------------------
    if (tid < 16) {
      int m = tid;
      long g = ((long)(b0 + m) * 200 + t) * 128;
      float s = 0.0f;
      for (int h = 0; h < 128; ++h) s += kkkF[m * 128 + h] * target_data[g + h];
      logits[(long)(b0 + m) * 200 + t] = s;
    }
  }
}

__global__ void final_kernel(const float* __restrict__ logits,
                             const int* __restrict__ tidx,
                             float* __restrict__ out, int nsel)
{
  int i = blockIdx.x * blockDim.x + threadIdx.x;
  if (i >= nsel) return;
  float v = logits[tidx[i]];
  out[i] = 1.0f / (1.0f + __expf(-v));
  out[nsel + i] = v;
}

// ---------------------------------------------------------------------------
extern "C" void kernel_launch(void* const* d_in, const int* in_sizes, int n_in,
                              void* d_out, int out_size, void* d_ws, size_t ws_size,
                              hipStream_t stream)
{
  (void)n_in; (void)out_size; (void)ws_size;
  const int* input_p      = (const int*)d_in[0];
  const int* target_p     = (const int*)d_in[1];
  const int* input_sd     = (const int*)d_in[2];
  const int* input_d      = (const int*)d_in[3];
  const float* input_kc   = (const float*)d_in[4];
  const int* x_answer     = (const int*)d_in[5];
  const int* target_sd    = (const int*)d_in[6];
  const int* target_d     = (const int*)d_in[7];
  const float* target_kc  = (const float*)d_in[8];
  const int* target_index = (const int*)d_in[9];
  const float* problem_embed = (const float*)d_in[10];
  const float* sd_embed   = (const float*)d_in[11];
  const float* diff_embed = (const float*)d_in[12];
  const float* answer_embed = (const float*)d_in[13];
  const float* skill_W    = (const float*)d_in[14];
  const float* skill_b    = (const float*)d_in[15];
  const float* tskill_W   = (const float*)d_in[16];
  const float* tskill_b   = (const float*)d_in[17];
  const float* Wl = (const float*)d_in[18];
  const float* bl = (const float*)d_in[19];
  const float* Wc = (const float*)d_in[20];
  const float* bc = (const float*)d_in[21];
  const float* Wo = (const float*)d_in[22];
  const float* bo = (const float*)d_in[23];
  const float* Wq = (const float*)d_in[24];
  const float* bq = (const float*)d_in[25];
  const float* W1 = (const float*)d_in[26];
  const float* b1 = (const float*)d_in[27];
  const float* Wip = (const float*)d_in[28];
  const float* bip = (const float*)d_in[29];
  const float* Wtp = (const float*)d_in[30];
  const float* btp = (const float*)d_in[31];
  const float* kw  = (const float*)d_in[32];

  const long MBT = 512L * 200L;   // 102400 (b*T rows)
  const int NPe = 53092, NDe = 1011;

  // workspace carve-out (floats), ~256 MB total
  float* ws = (float*)d_ws;
  float* inputd  = ws;                         // MBT*128
  float* targetd = inputd  + MBT * 128;
  float* prel    = targetd + MBT * 128;
  float* prec    = prel    + MBT * 128;
  float* P_ip    = prec    + MBT * 128;        // NPe*128
  float* P_tp    = P_ip    + (long)NPe * 128;
  float* SD_ip   = P_tp    + (long)NPe * 128;  // NDe*128 each
  float* D_ip    = SD_ip   + (long)NDe * 128;
  float* SD_tp   = D_ip    + (long)NDe * 128;
  float* D_tp    = SD_tp   + (long)NDe * 128;
  float* SD1     = D_tp    + (long)NDe * 128;
  float* D1t     = SD1     + (long)NDe * 128;
  float* C1      = D1t     + (long)NDe * 128;  // 128*265
  float* C2      = C1 + 128 * 265;
  float* ANSo    = C2 + 128 * 265;             // 2*128 each
  float* ANSq    = ANSo + 256;
  float* ANS1    = ANSq + 256;
  float* cvec_ip = ANS1 + 256;
  float* cvec_tp = cvec_ip + 128;
  float* logits  = cvec_tp + 128;              // MBT

  // 1) tiny constant tables
  misc_kernel<<<1, 128, 0, stream>>>(Wip, bip, skill_b, Wtp, btp, tskill_b,
                                     Wo, bo, Wq, bq, W1, b1, answer_embed,
                                     cvec_ip, cvec_tp, ANSo, ANSq, ANS1);

  // 2) C1 = Wip_sk @ skill_W (128 x 265), C2 = Wtp_sk @ tskill_W
  gemm_wmma<17, false><<<1, 256, 0, stream>>>(Wip + 128, 128, 128, 512,
                                              skill_W, 265, 265, nullptr, 1.0f, C1, 265);
  gemm_wmma<17, false><<<1, 256, 0, stream>>>(Wtp + 128, 128, 128, 512,
                                              tskill_W, 265, 265, nullptr, 1.0f, C2, 265);

  // 3) gather tables through projection columns (bias folded into P_*)
  auto blocksM = [](long M) { return (int)(((M + 15) / 16 + 7) / 8); };
  gemm_wmma<8, true><<<blocksM(NPe), 256, 0, stream>>>(problem_embed, NPe, 128, 128,
                                                       Wip, 512, 128, cvec_ip, 1.0f, P_ip, 128);
  gemm_wmma<8, true><<<blocksM(NPe), 256, 0, stream>>>(problem_embed, NPe, 128, 128,
                                                       Wtp, 512, 128, cvec_tp, 1.0f, P_tp, 128);
  gemm_wmma<8, true><<<blocksM(NDe), 256, 0, stream>>>(sd_embed, NDe, 128, 128,
                                                       Wip + 256, 512, 128, nullptr, 1.0f, SD_ip, 128);
  gemm_wmma<8, true><<<blocksM(NDe), 256, 0, stream>>>(diff_embed, NDe, 128, 128,
                                                       Wip + 384, 512, 128, nullptr, 1.0f, D_ip, 128);
  gemm_wmma<8, true><<<blocksM(NDe), 256, 0, stream>>>(sd_embed, NDe, 128, 128,
                                                       Wtp + 256, 512, 128, nullptr, 1.0f, SD_tp, 128);
  gemm_wmma<8, true><<<blocksM(NDe), 256, 0, stream>>>(diff_embed, NDe, 128, 128,
                                                       Wtp + 384, 512, 128, nullptr, 1.0f, D_tp, 128);
  gemm_wmma<8, true><<<blocksM(NDe), 256, 0, stream>>>(sd_embed, NDe, 128, 128,
                                                       W1 + 256, 512, 128, nullptr, 1.0f, SD1, 128);
  gemm_wmma<8, true><<<blocksM(NDe), 256, 0, stream>>>(diff_embed, NDe, 128, 128,
                                                       W1 + 384, 512, 128, nullptr, 1.0f, D1t, 128);

  // 4) main K=265 GEMMs: inputd = input_kc @ C1.T ; targetd = target_kc @ C2.T
  gemm_wmma<8, true><<<blocksM(MBT), 256, 0, stream>>>(input_kc, (int)MBT, 265, 265,
                                                       C1, 265, 128, nullptr, 1.0f, inputd, 128);
  gemm_wmma<8, true><<<blocksM(MBT), 256, 0, stream>>>(target_kc, (int)MBT, 265, 265,
                                                       C2, 265, 128, nullptr, 1.0f, targetd, 128);

  // 5) add gathered embedding projections -> full input_data / target_data
  int gblocks = (int)((MBT * 128 + 255) / 256);
  gather_add_kernel<<<gblocks, 256, 0, stream>>>(inputd, P_ip, SD_ip, D_ip,
                                                 input_p, input_sd, input_d, MBT);
  gather_add_kernel<<<gblocks, 256, 0, stream>>>(targetd, P_tp, SD_tp, D_tp,
                                                 target_p, target_sd, target_d, MBT);

  // 6) prel = bl - input_data@Wl.T ; prec = bc - input_data@Wc.T
  gemm_wmma<8, true><<<blocksM(MBT), 256, 0, stream>>>(inputd, (int)MBT, 128, 128,
                                                       Wl, 128, 128, bl, -1.0f, prel, 128);
  gemm_wmma<8, true><<<blocksM(MBT), 256, 0, stream>>>(inputd, (int)MBT, 128, 128,
                                                       Wc, 128, 128, bc, -1.0f, prec, 128);

  // 7) recurrence: 32 workgroups x 1 wave, 180.5 KB dynamic LDS each
  const size_t rnn_smem = (size_t)(5 * 16384 + 2048 + 2048) * 2 + 2048 * 4 + 64 * 4;
  rnn_wmma<<<32, 32, rnn_smem, stream>>>(Wl, Wc, Wo, Wq, W1, prel, prec,
                                         ANSo, ANSq, ANS1, SD1, D1t,
                                         x_answer, input_sd, input_d,
                                         targetd, kw, logits);

  // 8) final gather + sigmoid (tuple output: [sigmoid(sel), sel])
  int nsel = in_sizes[9];
  final_kernel<<<(nsel + 255) / 256, 256, 0, stream>>>(logits, target_index,
                                                       (float*)d_out, nsel);
}
